// Quant_Conv2d_Int_43877385896253
// MI455X (gfx1250) — 8-bit matrix operand order repaired and checked by reading, compiled, not run on hardware
//
#include <hip/hip_runtime.h>

constexpr int Bn = 16, Cc = 128, Hh = 56, Wd = 56, Co = 128;
constexpr int Ff = Cc * 9;
constexpr int Ll = Hh * Wd;
constexpr int Mm = Bn * Ll;
constexpr int XN = Bn * Cc * Hh * Wd;
constexpr int XBLOCKS = 1024;
constexpr int MT = Mm / 16;

constexpr size_t OFF_SCALARS = 0;
constexpr size_t OFF_SCALE_W = 256;
constexpr size_t OFF_ZP_W    = 768;
constexpr size_t OFF_SUM_QW  = 1280;
constexpr size_t OFF_QB      = 1792;
constexpr size_t OFF_XPART   = 2304;
constexpr size_t OFF_RPART   = 10496;
constexpr size_t OFF_SUMQX   = 35584;
constexpr size_t OFF_QW      = 236288;
constexpr size_t OFF_QX      = 383744;
constexpr size_t OFF_RES     = 58186496;


typedef __attribute__((ext_vector_type(8))) int v8i_t;

__device__ __forceinline__ float q8clip(float v) {
    return fminf(fmaxf(rintf(v), -128.0f), 127.0f);
}

__global__ void k_wquant(const float* __restrict__ w, float* __restrict__ scale_w,
                         float* __restrict__ zp_w, float* __restrict__ sum_qw,
                         signed char* __restrict__ qw) {
    const int o = blockIdx.x, t = threadIdx.x;
    const float* row = w + (size_t)o * Ff;
    float lmin = 1e30f, lmax = -1e30f;
    for (int j = t; j < Ff; j += 128) {
        float v = row[j];
        lmin = fminf(lmin, v); lmax = fmaxf(lmax, v);
    }
    __shared__ float smin[128], smax[128];
    smin[t] = lmin; smax[t] = lmax; __syncthreads();
    for (int s = 64; s > 0; s >>= 1) {
        if (t < s) { smin[t] = fminf(smin[t], smin[t + s]); smax[t] = fmaxf(smax[t], smax[t + s]); }
        __syncthreads();
    }
    const float wmin = smin[0], wmax = smax[0];
    const float sc = 255.0f / fmaxf(wmax - wmin, 1e-8f);
    const float zp = rintf(sc * wmin) + 128.0f;
    __syncthreads();
    float lsum = 0.0f;
    for (int j = t; j < Ff; j += 128) {
        float q = q8clip(sc * row[j] - zp);
        qw[(size_t)o * Ff + j] = (signed char)q;
        lsum += q;
    }
    smin[t] = lsum; __syncthreads();
    for (int s = 64; s > 0; s >>= 1) { if (t < s) smin[t] += smin[t + s]; __syncthreads(); }
    if (t == 0) { scale_w[o] = sc; zp_w[o] = zp; sum_qw[o] = smin[0]; }
}

__global__ void k_xminmax(const float* __restrict__ x, float* __restrict__ part) {
    float lmin = 0.0f, lmax = 0.0f;
    for (int i = blockIdx.x * blockDim.x + threadIdx.x; i < XN; i += gridDim.x * blockDim.x) {
        float v = x[i];
        lmin = fminf(lmin, v); lmax = fmaxf(lmax, v);
    }
    __shared__ float smin[256], smax[256];
    int t = threadIdx.x;
    smin[t] = lmin; smax[t] = lmax; __syncthreads();
    for (int s = 128; s > 0; s >>= 1) {
        if (t < s) { smin[t] = fminf(smin[t], smin[t + s]); smax[t] = fmaxf(smax[t], smax[t + s]); }
        __syncthreads();
    }
    if (t == 0) { part[2 * blockIdx.x] = smin[0]; part[2 * blockIdx.x + 1] = smax[0]; }
}

__global__ void k_xfin(const float* __restrict__ part, float* __restrict__ scalars,
                       const float* __restrict__ scale_w, const float* __restrict__ bias,
                       float* __restrict__ qb) {
    const int t = threadIdx.x;
    float lmin = 0.0f, lmax = 0.0f;
    for (int i = t; i < XBLOCKS; i += 128) {
        lmin = fminf(lmin, part[2 * i]); lmax = fmaxf(lmax, part[2 * i + 1]);
    }
    __shared__ float smin[128], smax[128];
    smin[t] = lmin; smax[t] = lmax; __syncthreads();
    for (int s = 64; s > 0; s >>= 1) {
        if (t < s) { smin[t] = fminf(smin[t], smin[t + s]); smax[t] = fmaxf(smax[t], smax[t + s]); }
        __syncthreads();
    }
    const float sc = 255.0f / fmaxf(smax[0] - smin[0], 1e-8f);
    const float zp = rintf(sc * smin[0]) + 128.0f;
    if (t == 0) { scalars[2] = sc; scalars[3] = zp; }
    qb[t] = rintf(scale_w[t] * sc * bias[t]);
}

__global__ void k_qpatch(const float* __restrict__ x, const float* __restrict__ scalars,
                         signed char* __restrict__ qx, float* __restrict__ sum_qx) {
    const int m = blockIdx.x;
    const int c = threadIdx.x;
    const float sc = scalars[2], zp = scalars[3];
    const int b = m / Ll, l = m % Ll;
    const int oh = l / Wd, ow = l % Wd;
    const float* xc = x + ((size_t)(b * Cc + c)) * (Hh * Wd);
    signed char* dst = qx + (size_t)m * Ff + c * 9;
    float lsum = 0.0f;
#pragma unroll
    for (int kh = 0; kh < 3; ++kh) {
        const int h = oh + kh - 1;
#pragma unroll
        for (int kw = 0; kw < 3; ++kw) {
            const int wc = ow + kw - 1;
            float v = (h >= 0 && h < Hh && wc >= 0 && wc < Wd) ? xc[h * Wd + wc] : 0.0f;
            float q = q8clip(sc * v - zp);
            dst[kh * 3 + kw] = (signed char)q;
            lsum += q;
        }
    }
    __shared__ float ssum[128];
    ssum[c] = lsum; __syncthreads();
    for (int s = 64; s > 0; s >>= 1) { if (c < s) ssum[c] += ssum[c + s]; __syncthreads(); }
    if (c == 0) sum_qx[m] = ssum[0];
}

__global__ void k_gemm(const signed char* __restrict__ qx, const signed char* __restrict__ qw,
                       const float* __restrict__ zp_w, const float* __restrict__ sum_qw,
                       const float* __restrict__ qb, const float* __restrict__ sum_qx,
                       const float* __restrict__ scalars,
                       float* __restrict__ res, float* __restrict__ rpart) {
    const int m0   = blockIdx.x * 16;
    const int wv   = threadIdx.x >> 5;
    const int lane = threadIdx.x & 31;
    const int n0   = wv * 16;
    const int ln   = lane & 15;
    const int g    = lane >> 4;
    const signed char* arow = qx + (size_t)(m0 + ln) * Ff;
    const signed char* brow = qw + (size_t)(n0 + ln) * Ff;

    v8i_t acc = {0, 0, 0, 0, 0, 0, 0, 0};
    for (int k0 = 0; k0 < Ff; k0 += 64) {
        if (k0 + 64 < Ff) __builtin_prefetch(arow + k0 + 64 + 8 * g, 0, 1);
        int2 a0 = *(const int2*)(arow + k0 + 8 * g);
        int2 a1 = *(const int2*)(arow + k0 + 8 * g + 16);
        int2 a2 = *(const int2*)(arow + k0 + 8 * g + 32);
        int2 a3 = *(const int2*)(arow + k0 + 8 * g + 48);
        v8i_t a = {a0.x, a0.y, a1.x, a1.y, a2.x, a2.y, a3.x, a3.y};
        int2 b0 = *(const int2*)(brow + k0 + 8 * g);
        int2 b1 = *(const int2*)(brow + k0 + 8 * g + 16);
        int2 b2 = *(const int2*)(brow + k0 + 8 * g + 32);
        int2 b3 = *(const int2*)(brow + k0 + 8 * g + 48);
        v8i_t b = {b0.x, b0.y, b1.x, b1.y, b2.x, b2.y, b3.x, b3.y};
        acc = __builtin_amdgcn_wmma_i32_16x16x64_iu8(true, a, true, b, acc, false, false);
    }

    const int o = n0 + ln;
    const float zpw = zp_w[o];
    const float zpx = scalars[3];
    const float base = zpx * sum_qw[o] + (float)Ff * zpw * zpx + qb[o];
    float lmin = 1e30f, lmax = -1e30f;
#pragma unroll
    for (int v = 0; v < 8; ++v) {
        const int mg = m0 + v + 8 * g;
        float r = (float)acc[v] + zpw * sum_qx[mg] + base;
        res[(size_t)mg * Co + o] = r;
        lmin = fminf(lmin, r); lmax = fmaxf(lmax, r);
    }
    __shared__ float smin[256], smax[256];
    const int t = threadIdx.x;
    smin[t] = lmin; smax[t] = lmax; __syncthreads();
    for (int s = 128; s > 0; s >>= 1) {
        if (t < s) { smin[t] = fminf(smin[t], smin[t + s]); smax[t] = fmaxf(smax[t], smax[t + s]); }
        __syncthreads();
    }
    if (t == 0) { rpart[2 * blockIdx.x] = smin[0]; rpart[2 * blockIdx.x + 1] = smax[0]; }
}

__global__ void k_rfin(const float* __restrict__ rpart, float* __restrict__ scalars) {
    const int t = threadIdx.x;
    float lmin = 1e30f, lmax = -1e30f;
    for (int i = t; i < MT; i += 256) {
        lmin = fminf(lmin, rpart[2 * i]); lmax = fmaxf(lmax, rpart[2 * i + 1]);
    }
    __shared__ float smin[256], smax[256];
    smin[t] = lmin; smax[t] = lmax; __syncthreads();
    for (int s = 128; s > 0; s >>= 1) {
        if (t < s) { smin[t] = fminf(smin[t], smin[t + s]); smax[t] = fmaxf(smax[t], smax[t + s]); }
        __syncthreads();
    }
    if (t == 0) {
        float sc = 255.0f / fmaxf(smax[0] - smin[0], 1e-8f);
        scalars[4] = smin[0]; scalars[5] = smax[0];
        scalars[6] = sc; scalars[7] = rintf(sc * smin[0]) + 128.0f;
    }
}

__global__ void k_out(const float* __restrict__ res, const float* __restrict__ scalars,
                      const float* __restrict__ scale_w, float* __restrict__ out) {
    const float sr = scalars[6], zpr = scalars[7], sx = scalars[2];
    const int total = Bn * Co * Ll;
    for (int idx = blockIdx.x * blockDim.x + threadIdx.x; idx < total; idx += gridDim.x * blockDim.x) {
        const int b = idx / (Co * Ll);
        const int rem = idx % (Co * Ll);
        const int o = rem / Ll;
        const int l = rem % Ll;
        const float r = res[(size_t)(b * Ll + l) * Co + o];
        const float q = q8clip(sr * r - zpr);
        out[idx] = (q + zpr) / sr / (scale_w[o] * sx);
    }
}

extern "C" void kernel_launch(void* const* d_in, const int* in_sizes, int n_in,
                              void* d_out, int out_size, void* d_ws, size_t ws_size,
                              hipStream_t stream) {
    (void)in_sizes; (void)n_in; (void)out_size; (void)ws_size;
    const float* x      = (const float*)d_in[0];
    const float* weight = (const float*)d_in[1];
    const float* bias   = (const float*)d_in[2];
    float* out = (float*)d_out;

    char* ws = (char*)d_ws;
    float*       scalars = (float*)(ws + OFF_SCALARS);
    float*       scale_w = (float*)(ws + OFF_SCALE_W);
    float*       zp_w    = (float*)(ws + OFF_ZP_W);
    float*       sum_qw  = (float*)(ws + OFF_SUM_QW);
    float*       qb      = (float*)(ws + OFF_QB);
    float*       xpart   = (float*)(ws + OFF_XPART);
    float*       rpart   = (float*)(ws + OFF_RPART);
    float*       sum_qx  = (float*)(ws + OFF_SUMQX);
    signed char* qw      = (signed char*)(ws + OFF_QW);
    signed char* qx      = (signed char*)(ws + OFF_QX);
    float*       res     = (float*)(ws + OFF_RES);

    k_wquant<<<Co, 128, 0, stream>>>(weight, scale_w, zp_w, sum_qw, qw);
    k_xminmax<<<XBLOCKS, 256, 0, stream>>>(x, xpart);
    k_xfin<<<1, 128, 0, stream>>>(xpart, scalars, scale_w, bias, qb);
    k_qpatch<<<Mm, 128, 0, stream>>>(x, scalars, qx, sum_qx);
    k_gemm<<<MT, 256, 0, stream>>>(qx, qw, zp_w, sum_qw, qb, sum_qx, scalars, res, rpart);
    k_rfin<<<1, 256, 0, stream>>>(rpart, scalars);
    k_out<<<4096, 256, 0, stream>>>(res, scalars, scale_w, out);
}
